// RRE_GNN_raw_33466385170788
// MI455X (gfx1250) — matrix operand order repaired and checked by reading, not compiled, not run on hardware
//
#include <hip/hip_runtime.h>
#include <hip/hip_bf16.h>
#include <math.h>

typedef __attribute__((ext_vector_type(16))) _Float16 v16h;
typedef __attribute__((ext_vector_type(8)))  float    v8f;

#define WAVES 4
#define HS_ST 132      // padded f32 row stride for 16x128 tiles
#define RH_ST 136      // padded f16 row stride
#define DD 128

__device__ __forceinline__ v8f wmma_f16(v16h a, v16h b, v8f c) {
  return __builtin_amdgcn_wmma_f32_16x16x32_f16(false, a, false, b, (short)0, c,
                                                false, false);
}

// Fast transcendentals: map onto v_exp_f32 / v_rcp_f32 (TRANS pipe, co-executes
// with the matrix pipe on CDNA5) instead of libm polynomial expansions.
__device__ __forceinline__ float fast_exp(float x) { return __expf(x); }
__device__ __forceinline__ float fast_sigmoid(float x) {
  return 1.f / (1.f + __expf(-x));
}
__device__ __forceinline__ float fast_tanh(float x) {
  x = fminf(15.f, fmaxf(-15.f, x));
  float t = __expf(2.f * x);
  return (t - 1.f) / (t + 1.f);
}

// Build an A-fragment (16x32 f16, M=rows of tile, K=feature dim) from an f32
// LDS tile with row stride HS_ST. ISA layout: lanes<16 hold K {0..7,16..23},
// lanes>=16 hold K {8..15,24..31} (offset by 32*ks).
__device__ __forceinline__ v16h ldA_f32(const float* tile, int ks, int lane) {
  int m  = lane & 15;
  int kb = (lane < 16) ? 0 : 8;
  const float* r = tile + m * HS_ST + ks * 32 + kb;
  v16h a;
#pragma unroll
  for (int i = 0; i < 8; ++i) {
    a[i]     = (_Float16)r[i];
    a[i + 8] = (_Float16)r[16 + i];
  }
  return a;
}

__device__ __forceinline__ v16h ldA_h(const _Float16* tile, int ks, int lane) {
  int m  = lane & 15;
  int kb = (lane < 16) ? 0 : 8;
  const _Float16* r = tile + m * RH_ST + ks * 32 + kb;
  v16h a;
#pragma unroll
  for (int i = 0; i < 8; ++i) { a[i] = r[i]; a[i + 8] = r[16 + i]; }
  return a;
}

// B fragments pre-shuffled to fragment-major: one contiguous 32B load per lane.
__device__ __forceinline__ v16h ldB_frag(const _Float16* frag, int ks, int nt,
                                         int lane) {
  return *(const v16h*)(frag + ((((ks * 8 + nt) * 32) + lane) << 4));
}

// ---------------- prep kernels ----------------

// Shuffle a 128x128 f32 matrix (starting at row_off) into fragment-major f16.
// dst index d = (((ks*8+nt)*32+lane)*16+e); B layout: lanes<16 K {0..7,16..23},
// lanes>=16 K {8..15,24..31} (within 32-wide k-step); column n = 16*nt + lane%16.
__global__ void prep_frag_kernel(const float* __restrict__ src, int row_off,
                                 _Float16* __restrict__ dst) {
  int d = blockIdx.x * blockDim.x + threadIdx.x;
  if (d >= DD * DD) return;
  int e    = d & 15;
  int lane = (d >> 4) & 31;
  int nt   = (d >> 9) & 7;
  int ks   = d >> 12;
  int K = 32 * ks + ((lane & 16) ? 8 : 0) + (e & 7) + ((e & 8) ? 16 : 0);
  int n = 16 * nt + (lane & 15);
  dst[d] = (_Float16)src[(long)(K + row_off) * DD + n];
}

__global__ void zero_kernel(float* agg, float* sum_exp, int* ng, int N) {
  long i      = (long)blockIdx.x * blockDim.x + threadIdx.x;
  long stride = (long)gridDim.x * blockDim.x;
  long tot    = (long)N * DD;
  for (long j = i; j < tot; j += stride) agg[j] = 0.f;
  for (long j = i; j < N; j += stride) { sum_exp[j] = 0.f; ng[j] = 0; }
}

__global__ void hq_kernel(const float* __restrict__ rela,
                          const int* __restrict__ q_rel,
                          float* __restrict__ hq, int NQ) {
  int i = blockIdx.x * blockDim.x + threadIdx.x;
  if (i < NQ * DD) hq[i] = rela[(long)q_rel[i >> 7] * DD + (i & 127)];
}

__global__ void ngroup_kernel(const int* __restrict__ edges, int* ng, int E) {
  int e = blockIdx.x * blockDim.x + threadIdx.x;
  if (e < E) ng[edges[(long)e * 6 + 5]] = edges[(long)e * 6 + 0];
}

// out[M x 128] = A[M x 128] @ B(128x128, frag-major) + bias (optional)
__global__ void gemm128_kernel(const float* __restrict__ A, int M,
                               const _Float16* __restrict__ Bfrag,
                               const float* __restrict__ bias,
                               float* __restrict__ out) {
  __shared__ float at[WAVES][16 * HS_ST];
  int wave = threadIdx.x >> 5, lane = threadIdx.x & 31;
  float* T = at[wave];
  int r0 = (blockIdx.x * WAVES + wave) * 16;
  for (int i = lane; i < 16 * 32; i += 32) {
    int m = i >> 5, c4 = (i & 31) * 4;
    int row  = r0 + m;
    float4 v = make_float4(0.f, 0.f, 0.f, 0.f);
    if (row < M) v = *(const float4*)(A + (long)row * DD + c4);
    *(float4*)&T[m * HS_ST + c4] = v;
  }
  __syncthreads();
  int nloc = lane & 15, moff = (lane < 16) ? 0 : 8;
  for (int nt = 0; nt < 8; ++nt) {
    v8f acc = {};
#pragma unroll
    for (int ks = 0; ks < 4; ++ks)
      acc = wmma_f16(ldA_f32(T, ks, lane), ldB_frag(Bfrag, ks, nt, lane), acc);
    int   n  = nt * 16 + nloc;
    float bv = bias ? bias[n] : 0.f;
#pragma unroll
    for (int v = 0; v < 8; ++v) {
      int row = r0 + v + moff;
      if (row < M) out[(long)row * DD + n] = acc[v] + bv;
    }
  }
}

// ---------------- main per-edge kernel ----------------

struct __align__(16) WaveScratch {
  float    hs[16 * HS_ST];   // gathered hidden[sub] tile (f32)
  float    zt[16 * HS_ST];   // z gate
  float    msg[16 * HS_ST];  // message
  _Float16 rh[16 * RH_ST];   // r * h_s (f16, A-operand for Uh GEMM)
  float    expt[16];         // exp(attn) per edge
  int er[16], erel[16], esub[16], eobj[16];
};

__global__ void edge_kernel(const int* __restrict__ edges, int E,
                            const float* __restrict__ hidden,
                            const float* __restrict__ preRelZ,
                            const float* __restrict__ preRelR,
                            const float* __restrict__ preRelH,
                            const float* __restrict__ preQZ,
                            const float* __restrict__ preQR,
                            const float* __restrict__ preQH,
                            const float* __restrict__ preQA,
                            const _Float16* __restrict__ fUz,
                            const _Float16* __restrict__ fUr,
                            const _Float16* __restrict__ fUh,
                            const _Float16* __restrict__ fWs,
                            const float* __restrict__ w_alpha,
                            const float* __restrict__ b_alpha,
                            float* __restrict__ sum_exp,
                            float* __restrict__ agg) {
  __shared__ WaveScratch S[WAVES];
  int wave = threadIdx.x >> 5, lane = threadIdx.x & 31;
  WaveScratch& W = S[wave];
  long e0 = ((long)blockIdx.x * WAVES + wave) * 16;

  if (lane < 16) {
    long e = e0 + lane;
    if (e >= E) e = E - 1;  // clamp; stores are guarded below
    const int* ep = edges + e * 6;
    W.er[lane] = ep[0]; W.erel[lane] = ep[2];
    W.esub[lane] = ep[4]; W.eobj[lane] = ep[5];
  }
  __syncthreads();

  // gather h_s tile (16 rows x 128), float4 per lane-iteration
  for (int i = lane; i < 16 * 32; i += 32) {
    int m = i >> 5, c4 = (i & 31) * 4;
    float4 v = *(const float4*)(hidden + (long)W.esub[m] * DD + c4);
    *(float4*)&W.hs[m * HS_ST + c4] = v;
  }
  __syncthreads();

  int nloc = lane & 15, moff = (lane < 16) ? 0 : 8;

  // ---- gates z and r: h_s @ Uz, h_s @ Ur ----
  for (int nt = 0; nt < 8; ++nt) {
    v8f az = {}, ar = {};
#pragma unroll
    for (int ks = 0; ks < 4; ++ks) {
      v16h a = ldA_f32(W.hs, ks, lane);
      az = wmma_f16(a, ldB_frag(fUz, ks, nt, lane), az);
      ar = wmma_f16(a, ldB_frag(fUr, ks, nt, lane), ar);
    }
    int n = nt * 16 + nloc;
#pragma unroll
    for (int v = 0; v < 8; ++v) {
      int  m  = v + moff;
      long rl = W.erel[m], rq = W.er[m];
      float zv = fast_sigmoid(az[v] + preRelZ[rl * DD + n] + preQZ[rq * DD + n]);
      W.zt[m * HS_ST + n] = zv;
      float rv = fast_sigmoid(ar[v] + preRelR[rl * DD + n] + preQR[rq * DD + n]);
      W.rh[m * RH_ST + n] = (_Float16)(rv * W.hs[m * HS_ST + n]);
    }
  }
  __syncthreads();

  // ---- h_tilde = tanh((r*h_s)@Uh + pre) ; message ----
  for (int nt = 0; nt < 8; ++nt) {
    v8f ah = {};
#pragma unroll
    for (int ks = 0; ks < 4; ++ks)
      ah = wmma_f16(ldA_h(W.rh, ks, lane), ldB_frag(fUh, ks, nt, lane), ah);
    int n = nt * 16 + nloc;
#pragma unroll
    for (int v = 0; v < 8; ++v) {
      int  m  = v + moff;
      float ht = fast_tanh(ah[v] + preRelH[(long)W.erel[m] * DD + n] +
                           preQH[(long)W.er[m] * DD + n]);
      float zv  = W.zt[m * HS_ST + n];
      float hsv = W.hs[m * HS_ST + n];
      W.msg[m * HS_ST + n] = (1.f - zv) * hsv + zv * ht;
    }
  }
  __syncthreads();

  // ---- attention: sum_n relu(msg@Ws + preQA)[n] * w_alpha[n] ----
  float attn[8];
#pragma unroll
  for (int v = 0; v < 8; ++v) attn[v] = 0.f;
  for (int nt = 0; nt < 8; ++nt) {
    v8f aa = {};
#pragma unroll
    for (int ks = 0; ks < 4; ++ks)
      aa = wmma_f16(ldA_f32(W.msg, ks, lane), ldB_frag(fWs, ks, nt, lane), aa);
    int   n  = nt * 16 + nloc;
    float wa = w_alpha[n];
#pragma unroll
    for (int v = 0; v < 8; ++v) {
      int   m = v + moff;
      float t = aa[v] + preQA[(long)W.er[m] * DD + n];
      attn[v] += (t > 0.f ? t : 0.f) * wa;
    }
  }
#pragma unroll
  for (int s = 1; s < 16; s <<= 1)
#pragma unroll
    for (int v = 0; v < 8; ++v) attn[v] += __shfl_xor(attn[v], s, 32);
  float ba = b_alpha[0];
  if (nloc == 0) {
#pragma unroll
    for (int v = 0; v < 8; ++v) W.expt[v + moff] = fast_exp(attn[v] + ba);
  }
  __syncthreads();

  // ---- segment-sum scatter ----
  if (nloc == 0) {
#pragma unroll
    for (int v = 0; v < 8; ++v) {
      int m = v + moff;
      if (e0 + m < E) atomicAdd(&sum_exp[W.eobj[m]], W.expt[m]);
    }
  }
  for (int i = lane; i < 16 * DD; i += 32) {
    int m = i >> 7, n = i & 127;
    if (e0 + m < E)
      atomicAdd(&agg[(long)W.eobj[m] * DD + n], W.expt[m] * W.msg[m * HS_ST + n]);
  }
}

// ---------------- finalize: hidden_new = relu((agg/sum)@W_h); h_n_qr ----------------

__global__ void finalize_kernel(const float* __restrict__ agg,
                                const float* __restrict__ sum_exp,
                                const _Float16* __restrict__ fWh,
                                const int* __restrict__ ng,
                                const float* __restrict__ hq,
                                float* __restrict__ out1,
                                float* __restrict__ out2, int N) {
  __shared__ float at[WAVES][16 * HS_ST];
  int wave = threadIdx.x >> 5, lane = threadIdx.x & 31;
  float* T = at[wave];
  int r0 = (blockIdx.x * WAVES + wave) * 16;
  for (int i = lane; i < 16 * DD; i += 32) {
    int m = i >> 7, n = i & 127;
    int row = r0 + m;
    float v = 0.f;
    if (row < N) v = agg[(long)row * DD + n] / sum_exp[row];
    T[m * HS_ST + n] = v;
  }
  __syncthreads();
  int nloc = lane & 15, moff = (lane < 16) ? 0 : 8;
  for (int nt = 0; nt < 8; ++nt) {
    v8f acc = {};
#pragma unroll
    for (int ks = 0; ks < 4; ++ks)
      acc = wmma_f16(ldA_f32(T, ks, lane), ldB_frag(fWh, ks, nt, lane), acc);
    int n = nt * 16 + nloc;
#pragma unroll
    for (int v = 0; v < 8; ++v) {
      int row = r0 + v + moff;
      if (row < N) out1[(long)row * DD + n] = acc[v] > 0.f ? acc[v] : 0.f;
    }
  }
  for (int i = lane; i < 16 * DD; i += 32) {
    int m = i >> 7, n = i & 127;
    int row = r0 + m;
    if (row < N) out2[(long)row * DD + n] = hq[(long)ng[row] * DD + n];
  }
}

// ---------------- host ----------------

extern "C" void kernel_launch(void* const* d_in, const int* in_sizes, int n_in,
                              void* d_out, int out_size, void* d_ws,
                              size_t ws_size, hipStream_t stream) {
  (void)n_in; (void)out_size; (void)ws_size;
  const float* hidden = (const float*)d_in[0];
  const float* rela   = (const float*)d_in[1];
  const float* Wz     = (const float*)d_in[2];
  const float* Uz     = (const float*)d_in[3];
  const float* bz     = (const float*)d_in[4];
  const float* Wr     = (const float*)d_in[5];
  const float* Ur     = (const float*)d_in[6];
  const float* br     = (const float*)d_in[7];
  const float* Whh    = (const float*)d_in[8];
  const float* Uh     = (const float*)d_in[9];
  const float* bh     = (const float*)d_in[10];
  const float* Wsat   = (const float*)d_in[11];
  const float* Wqr    = (const float*)d_in[12];
  const float* bqr    = (const float*)d_in[13];
  const float* walpha = (const float*)d_in[14];
  const float* balpha = (const float*)d_in[15];
  const float* Whmat  = (const float*)d_in[16];
  const int*   q_rel  = (const int*)d_in[17];
  const int*   edges  = (const int*)d_in[18];

  int N   = in_sizes[0] / DD;
  int NRE = in_sizes[1] / DD;
  int NQ  = in_sizes[17];
  int E   = in_sizes[18] / 6;

  char*  w   = (char*)d_ws;
  size_t off = 0;
  auto alloc = [&](size_t bytes) -> char* {
    char* p = w + off;
    off = (off + bytes + 255) & ~(size_t)255;
    return p;
  };

  // frag order: 0:Uz 1:Ur 2:Uh 3:Ws_attn 4:W_h 5:WzT 6:WzB 7:WrT 8:WrB 9:WhhT 10:WhhB 11:Wqr
  _Float16* frag[12];
  for (int i = 0; i < 12; ++i) frag[i] = (_Float16*)alloc((size_t)DD * DD * 2);
  float* hq      = (float*)alloc((size_t)NQ * DD * 4);
  float* preQZ   = (float*)alloc((size_t)NQ * DD * 4);
  float* preQR   = (float*)alloc((size_t)NQ * DD * 4);
  float* preQH   = (float*)alloc((size_t)NQ * DD * 4);
  float* preQA   = (float*)alloc((size_t)NQ * DD * 4);
  int NREp = ((NRE + 15) / 16) * 16;
  float* preRelZ = (float*)alloc((size_t)NREp * DD * 4);
  float* preRelR = (float*)alloc((size_t)NREp * DD * 4);
  float* preRelH = (float*)alloc((size_t)NREp * DD * 4);
  float* sum_exp = (float*)alloc((size_t)N * 4);
  int*   ng      = (int*)alloc((size_t)N * 4);
  float* agg     = (float*)alloc((size_t)N * DD * 4);

  float* out1 = (float*)d_out;
  float* out2 = out1 + (size_t)N * DD;

  zero_kernel<<<512, 256, 0, stream>>>(agg, sum_exp, ng, N);

  const float* fsrc[12] = {Uz, Ur, Uh, Wsat, Whmat, Wz, Wz, Wr, Wr, Whh, Whh, Wqr};
  const int    foff[12] = {0, 0, 0, 0, 0, 0, DD, 0, DD, 0, DD, 0};
  for (int i = 0; i < 12; ++i)
    prep_frag_kernel<<<(DD * DD + 255) / 256, 256, 0, stream>>>(fsrc[i], foff[i],
                                                                frag[i]);

  hq_kernel<<<(NQ * DD + 255) / 256, 256, 0, stream>>>(rela, q_rel, hq, NQ);

  int qb = (((NQ + 15) / 16) + WAVES - 1) / WAVES;
  gemm128_kernel<<<qb, 32 * WAVES, 0, stream>>>(hq, NQ, frag[6], nullptr, preQZ);
  gemm128_kernel<<<qb, 32 * WAVES, 0, stream>>>(hq, NQ, frag[8], nullptr, preQR);
  gemm128_kernel<<<qb, 32 * WAVES, 0, stream>>>(hq, NQ, frag[10], nullptr, preQH);
  gemm128_kernel<<<qb, 32 * WAVES, 0, stream>>>(hq, NQ, frag[11], bqr, preQA);

  int rb = (((NRE + 15) / 16) + WAVES - 1) / WAVES;
  gemm128_kernel<<<rb, 32 * WAVES, 0, stream>>>(rela, NRE, frag[5], bz, preRelZ);
  gemm128_kernel<<<rb, 32 * WAVES, 0, stream>>>(rela, NRE, frag[7], br, preRelR);
  gemm128_kernel<<<rb, 32 * WAVES, 0, stream>>>(rela, NRE, frag[9], bh, preRelH);

  ngroup_kernel<<<(E + 255) / 256, 256, 0, stream>>>(edges, ng, E);

  int eb = (((E + 15) / 16) + WAVES - 1) / WAVES;
  edge_kernel<<<eb, 32 * WAVES, 0, stream>>>(
      edges, E, hidden, preRelZ, preRelR, preRelH, preQZ, preQR, preQH, preQA,
      frag[0], frag[1], frag[2], frag[3], walpha, balpha, sum_exp, agg);

  int nb = (((N + 15) / 16) + WAVES - 1) / WAVES;
  finalize_kernel<<<nb, 32 * WAVES, 0, stream>>>(agg, sum_exp, frag[4], ng, hq,
                                                 out1, out2, N);
}
